// QPLEXSIWeight_4329327034500
// MI455X (gfx1250) — compile-verified, matrix operand order checked by reading, not run on hardware
//
#include <hip/hip_runtime.h>
#include <hip/hip_bf16.h>

typedef __attribute__((ext_vector_type(16))) _Float16 v16h;
typedef __attribute__((ext_vector_type(8)))  _Float16 v8h;
typedef __attribute__((ext_vector_type(8)))  float    v8f;

#define B_TOT   65536
#define S_DIM   256
#define A_N     8
#define K_N     10
#define H_DIM   64
#define AD_DIM  128
#define D_DIM   384

#define WG_ROWS 64          // rows of B per workgroup
#define NTHREADS 64         // 2 waves (wave32)
#define NWAVES  (NTHREADS / 32)
#define ROWS_PER_WAVE 32    // two 16-row M-tiles per wave -> B-fragment reuse x2
#define XS      392         // LDS stride for X tile (f16), multiple of 8, breaks bank aliasing
#define HS      72          // LDS stride for h tile (f16), multiple of 8

// ---------------------------------------------------------------------------
// Load a 16x32 f16 WMMA A/B fragment. Both A and B use the same per-lane
// layout (reduction dim contiguous in memory):
//   lanes 0-15 : row = lane,   v0..3 = K[k0..k0+7],  v4..7 = K[k0+16..k0+23]
//   lanes16-31 : row = lane-16,v0..3 = K[k0+8..+15], v4..7 = K[k0+24..+31]
// => two contiguous 16-byte loads per lane (ds_load_b128 / global_load_b128).
// ---------------------------------------------------------------------------
__device__ __forceinline__ v16h load_frag(const _Float16* __restrict__ p,
                                          int stride, int row, int k0, int lane) {
    const int ho = (lane >> 4) << 3;                 // 0 or 8
    const _Float16* q = p + row * stride + k0 + ho;
    union { v16h v; struct { v8h lo; v8h hi; } s; } u;
    u.s.lo = *(const v8h*)q;
    u.s.hi = *(const v8h*)(q + 16);
    return u.v;
}

// C tiles [2][16,64]: two M-tiles x W^T; every B fragment feeds 2 WMMAs.
template <int KD>
__device__ __forceinline__ void gemm4x2(const _Float16* __restrict__ A, int lda,
                                        const _Float16* __restrict__ W, int ldw,
                                        int lane, v8f c[2][4]) {
    const int mr = lane & 15;
    v8f z = {0.f, 0.f, 0.f, 0.f, 0.f, 0.f, 0.f, 0.f};
#pragma unroll
    for (int m = 0; m < 2; ++m)
#pragma unroll
        for (int t = 0; t < 4; ++t) c[m][t] = z;
    for (int k0 = 0; k0 < KD; k0 += 32) {
        v16h a0 = load_frag(A, lda, mr, k0, lane);
        v16h a1 = load_frag(A + 16 * lda, lda, mr, k0, lane);
#pragma unroll
        for (int t = 0; t < 4; ++t) {
            v16h b = load_frag(W, ldw, t * 16 + mr, k0, lane);
            c[0][t] = __builtin_amdgcn_wmma_f32_16x16x32_f16(
                false, a0, false, b, (short)0, c[0][t], false, false);
            c[1][t] = __builtin_amdgcn_wmma_f32_16x16x32_f16(
                false, a1, false, b, (short)0, c[1][t], false, false);
        }
    }
}

// C tiles [2][16,16] = two M-tiles x W3^T, W3 stored [16,64] (rows >= outdim zero)
__device__ __forceinline__ void gemm1x2(const _Float16* __restrict__ A, int lda,
                                        const _Float16* __restrict__ W,
                                        int lane, v8f c[2]) {
    const int mr = lane & 15;
    v8f z = {0.f, 0.f, 0.f, 0.f, 0.f, 0.f, 0.f, 0.f};
    c[0] = z; c[1] = z;
#pragma unroll
    for (int k0 = 0; k0 < H_DIM; k0 += 32) {
        v16h a0 = load_frag(A, lda, mr, k0, lane);
        v16h a1 = load_frag(A + 16 * lda, lda, mr, k0, lane);
        v16h b  = load_frag(W, H_DIM, mr, k0, lane);
        c[0] = __builtin_amdgcn_wmma_f32_16x16x32_f16(
            false, a0, false, b, (short)0, c[0], false, false);
        c[1] = __builtin_amdgcn_wmma_f32_16x16x32_f16(
            false, a1, false, b, (short)0, c[1], false, false);
    }
}

// C-tile layout: vgpr r, lane -> row = r + 8*(lane>>4), col = lane&15
__device__ __forceinline__ void bias_relu_store2(const v8f c[2][4],
                                                 const float* __restrict__ bias,
                                                 _Float16* __restrict__ Hout, int lane) {
    const int col = lane & 15, hi = lane >> 4;
#pragma unroll
    for (int t = 0; t < 4; ++t) {
        float bv = bias[t * 16 + col];
#pragma unroll
        for (int m = 0; m < 2; ++m)
#pragma unroll
            for (int r = 0; r < 8; ++r) {
                float v = c[m][t][r] + bv;
                Hout[(m * 16 + r + 8 * hi) * HS + t * 16 + col] =
                    (_Float16)(v > 0.f ? v : 0.f);
            }
    }
}

// ---------------------------------------------------------------------------
// Weight prep: fp32 -> f16, and pad layer-3 weights [K,out,H] -> [K,16,H]
// ---------------------------------------------------------------------------
__global__ void cvt_f32_f16(const float* __restrict__ src, _Float16* __restrict__ dst, int n) {
    int i = blockIdx.x * blockDim.x + threadIdx.x;
    if (i < n) dst[i] = (_Float16)src[i];
}

__global__ void cvt_pad_w3(const float* __restrict__ src, _Float16* __restrict__ dst, int outdim) {
    int i = blockIdx.x * blockDim.x + threadIdx.x;
    const int n = K_N * 16 * H_DIM;
    if (i >= n) return;
    int h = i % H_DIM;
    int o = (i / H_DIM) % 16;
    int k = i / (16 * H_DIM);
    dst[i] = (o < outdim) ? (_Float16)src[(k * outdim + o) * H_DIM + h] : (_Float16)0.f;
}

// ---------------------------------------------------------------------------
// Main fused kernel: per block, stage 64 rows of [states || actions] as f16 in
// LDS; each wave owns 32 rows (2 M-tiles) and runs all 3 MLPs for all K
// kernels via WMMA, reducing over K in registers.
// ---------------------------------------------------------------------------
__global__ __launch_bounds__(NTHREADS, 2)
void qplex_main(const float* __restrict__ states, const float* __restrict__ actions,
                const _Float16* __restrict__ wk1, const _Float16* __restrict__ wa1,
                const _Float16* __restrict__ wc1,
                const _Float16* __restrict__ wk2, const _Float16* __restrict__ wa2,
                const _Float16* __restrict__ wc2,
                const _Float16* __restrict__ wk3, const _Float16* __restrict__ wa3,
                const _Float16* __restrict__ wc3,
                const float* __restrict__ bk1, const float* __restrict__ bk2,
                const float* __restrict__ bk3,
                const float* __restrict__ ba1, const float* __restrict__ ba2,
                const float* __restrict__ ba3,
                const float* __restrict__ bc1, const float* __restrict__ bc2,
                const float* __restrict__ bc3,
                float* __restrict__ out) {
    __shared__ _Float16 Xs[WG_ROWS * XS];                     // 50176 B
    __shared__ _Float16 Hs[NWAVES * ROWS_PER_WAVE * HS];      //  9216 B (wave-private)

    const int tid = threadIdx.x;
    const int b0  = blockIdx.x * WG_ROWS;

    // Stage & convert this block's input tile: [64, 384] fp32 -> f16
    for (int idx = tid; idx < WG_ROWS * D_DIM; idx += NTHREADS) {
        int row = idx / D_DIM;
        int col = idx - row * D_DIM;
        float v = (col < S_DIM)
                      ? states[(size_t)(b0 + row) * S_DIM + col]
                      : actions[(size_t)(b0 + row) * AD_DIM + (col - S_DIM)];
        Xs[row * XS + col] = (_Float16)v;
    }
    __syncthreads();

    const int wave = tid >> 5, lane = tid & 31;
    const int col = lane & 15, hi = lane >> 4;
    const _Float16* Xw = Xs + (wave * ROWS_PER_WAVE) * XS;
    _Float16*       Hw = Hs + (wave * ROWS_PER_WAVE) * HS;

    float acc[2][8];
#pragma unroll
    for (int m = 0; m < 2; ++m)
#pragma unroll
        for (int r = 0; r < 8; ++r) acc[m][r] = 0.f;

#pragma unroll 1
    for (int k = 0; k < K_N; ++k) {
        v8f c1[2][4];
        v8f ck[2], ca[2], cc[2];

        // ---- key extractor: S -> H -> H -> 1 ----
        gemm4x2<S_DIM>(Xw, XS, wk1 + (size_t)k * H_DIM * S_DIM, S_DIM, lane, c1);
        bias_relu_store2(c1, bk1 + k * H_DIM, Hw, lane);
        gemm4x2<H_DIM>(Hw, HS, wk2 + (size_t)k * H_DIM * H_DIM, H_DIM, lane, c1);
        bias_relu_store2(c1, bk2 + k * H_DIM, Hw, lane);
        gemm1x2(Hw, HS, wk3 + (size_t)k * 16 * H_DIM, lane, ck);

        // ---- agents extractor: S -> H -> H -> A ----
        gemm4x2<S_DIM>(Xw, XS, wa1 + (size_t)k * H_DIM * S_DIM, S_DIM, lane, c1);
        bias_relu_store2(c1, ba1 + k * H_DIM, Hw, lane);
        gemm4x2<H_DIM>(Hw, HS, wa2 + (size_t)k * H_DIM * H_DIM, H_DIM, lane, c1);
        bias_relu_store2(c1, ba2 + k * H_DIM, Hw, lane);
        gemm1x2(Hw, HS, wa3 + (size_t)k * 16 * H_DIM, lane, ca);

        // ---- action extractor: S+AD -> H -> H -> A ----
        gemm4x2<D_DIM>(Xw, XS, wc1 + (size_t)k * H_DIM * D_DIM, D_DIM, lane, c1);
        bias_relu_store2(c1, bc1 + k * H_DIM, Hw, lane);
        gemm4x2<H_DIM>(Hw, HS, wc2 + (size_t)k * H_DIM * H_DIM, H_DIM, lane, c1);
        bias_relu_store2(c1, bc2 + k * H_DIM, Hw, lane);
        gemm1x2(Hw, HS, wc3 + (size_t)k * 16 * H_DIM, lane, cc);

        // ---- combine: acc += |key| * sigmoid(agents) * sigmoid(action) ----
        const float kbias = bk3[k];                     // [K,1]
        const float ab3v  = ba3[k * A_N + (col & 7)];   // [K,A]
        const float cb3v  = bc3[k * A_N + (col & 7)];
#pragma unroll
        for (int m = 0; m < 2; ++m)
#pragma unroll
            for (int r = 0; r < 8; ++r) {
                float kv = __shfl(ck[m][r], lane & 16, 32);  // broadcast col 0 per half
                kv = fabsf(kv + kbias) + 1e-10f;
                float ag = 1.f / (1.f + __expf(-(ca[m][r] + ab3v)));
                float ac = 1.f / (1.f + __expf(-(cc[m][r] + cb3v)));
                acc[m][r] += kv * ag * ac;
            }
    }

    if (col < A_N) {
#pragma unroll
        for (int m = 0; m < 2; ++m)
#pragma unroll
            for (int r = 0; r < 8; ++r)
                out[(size_t)(b0 + wave * ROWS_PER_WAVE + m * 16 + r + 8 * hi) * A_N + col] =
                    acc[m][r];
    }
}

// ---------------------------------------------------------------------------
extern "C" void kernel_launch(void* const* d_in, const int* in_sizes, int n_in,
                              void* d_out, int out_size, void* d_ws, size_t ws_size,
                              hipStream_t stream) {
    (void)in_sizes; (void)n_in; (void)out_size; (void)ws_size;
    const float* states  = (const float*)d_in[0];
    const float* actions = (const float*)d_in[1];
    const float* kW1 = (const float*)d_in[2];
    const float* kb1 = (const float*)d_in[3];
    const float* kW2 = (const float*)d_in[4];
    const float* kb2 = (const float*)d_in[5];
    const float* kW3 = (const float*)d_in[6];
    const float* kb3 = (const float*)d_in[7];
    const float* aW1 = (const float*)d_in[8];
    const float* ab1 = (const float*)d_in[9];
    const float* aW2 = (const float*)d_in[10];
    const float* ab2 = (const float*)d_in[11];
    const float* aW3 = (const float*)d_in[12];
    const float* ab3 = (const float*)d_in[13];
    const float* cW1 = (const float*)d_in[14];
    const float* cb1 = (const float*)d_in[15];
    const float* cW2 = (const float*)d_in[16];
    const float* cb2 = (const float*)d_in[17];
    const float* cW3 = (const float*)d_in[18];
    const float* cb3 = (const float*)d_in[19];

    // f16 weight arena in d_ws (~1.5 MB; L2-resident during the main kernel)
    _Float16* ws = (_Float16*)d_ws;
    size_t off = 0;
    _Float16* wk1 = ws + off; off += (size_t)K_N * H_DIM * S_DIM;
    _Float16* wa1 = ws + off; off += (size_t)K_N * H_DIM * S_DIM;
    _Float16* wc1 = ws + off; off += (size_t)K_N * H_DIM * D_DIM;
    _Float16* wk2 = ws + off; off += (size_t)K_N * H_DIM * H_DIM;
    _Float16* wa2 = ws + off; off += (size_t)K_N * H_DIM * H_DIM;
    _Float16* wc2 = ws + off; off += (size_t)K_N * H_DIM * H_DIM;
    _Float16* wk3 = ws + off; off += (size_t)K_N * 16 * H_DIM;
    _Float16* wa3 = ws + off; off += (size_t)K_N * 16 * H_DIM;
    _Float16* wc3 = ws + off; off += (size_t)K_N * 16 * H_DIM;

    auto cvt = [&](const float* s, _Float16* d, int n) {
        cvt_f32_f16<<<(n + 255) / 256, 256, 0, stream>>>(s, d, n);
    };
    cvt(kW1, wk1, K_N * H_DIM * S_DIM);
    cvt(aW1, wa1, K_N * H_DIM * S_DIM);
    cvt(cW1, wc1, K_N * H_DIM * D_DIM);
    cvt(kW2, wk2, K_N * H_DIM * H_DIM);
    cvt(aW2, wa2, K_N * H_DIM * H_DIM);
    cvt(cW2, wc2, K_N * H_DIM * H_DIM);
    const int n3 = K_N * 16 * H_DIM;
    cvt_pad_w3<<<(n3 + 255) / 256, 256, 0, stream>>>(kW3, wk3, 1);
    cvt_pad_w3<<<(n3 + 255) / 256, 256, 0, stream>>>(aW3, wa3, A_N);
    cvt_pad_w3<<<(n3 + 255) / 256, 256, 0, stream>>>(cW3, wc3, A_N);

    qplex_main<<<B_TOT / WG_ROWS, NTHREADS, 0, stream>>>(
        states, actions,
        wk1, wa1, wc1, wk2, wa2, wc2, wk3, wa3, wc3,
        kb1, kb2, kb3, ab1, ab2, ab3, cb1, cb2, cb3,
        (float*)d_out);
}
